// recursivefunc_87445534147416
// MI455X (gfx1250) — matrix operand order repaired and checked by reading, not compiled, not run on hardware
//
#include <hip/hip_runtime.h>
#include <math.h>

// ---------------------------------------------------------------------------
// Problem dimensions (fixed by the reference)
//   B=1024, LQ=20, LH=10, NI=36, H=1024, NINP=1024
// All GEMM dims are multiples of the block tile (128x128, BK=32) -> no edges.
// ---------------------------------------------------------------------------

typedef __attribute__((ext_vector_type(16))) _Float16 f16x16;
typedef __attribute__((ext_vector_type(8)))  _Float16 f16x8;
typedef __attribute__((ext_vector_type(4)))  _Float16 f16x4;
typedef __attribute__((ext_vector_type(8)))  float    f32x8;

#define BM 128
#define BN 128
#define BK 32
#define LDS_STRIDE 40   // halves per LDS row (32 data + 8 pad; 80B keeps 16B align)

// Branch-free tanh: prefer the CDNA5 hardware transcendental v_tanh_f32.
__device__ __forceinline__ float fast_tanh(float x)
{
#if __has_builtin(__builtin_amdgcn_tanhf)
    return __builtin_amdgcn_tanhf(x);
#else
    // branchless fallback: tanh(x) = 1 - 2/(exp(2x)+1); saturates correctly.
    const float e = __expf(2.0f * x);
    return 1.0f - 2.0f / (e + 1.0f);
#endif
}

// ---------------------------------------------------------------------------
// WMMA GEMM:  C[M,N] = op( BETA*C + A[M,K] @ W[K,N] + bias[N] )
// fp32 in/out, f16 WMMA compute with fp32 accumulation.
// 256 threads = 8 wave32s arranged 4(M) x 2(N); each wave owns a 32x64 tile
// built from 2x4 v_wmma_f32_16x16x32_f16 (8 WMMA per K-step).
// Global->register prefetch pipelines HBM latency under the WMMA burst.
// ---------------------------------------------------------------------------
template<int BETA, int TANH>
__global__ __launch_bounds__(256)
void wmma_gemm(const float* __restrict__ A, int lda,
               const float* __restrict__ W, int ldw,
               const float* __restrict__ bias,
               float* __restrict__ C, int ldc,
               int M, int N, int K)
{
    __shared__ _Float16 As[BM * LDS_STRIDE];   // [row m][k]   (row-major)
    __shared__ _Float16 Bs[BN * LDS_STRIDE];   // [col n][k]   (W transposed)

    const int tid  = threadIdx.x;
    const int m0   = blockIdx.y * BM;
    const int n0   = blockIdx.x * BN;
    const int lane = tid & 31;
    const int wid  = tid >> 5;
    const int wm   = wid >> 1;        // 0..3  (M direction, 32 rows each)
    const int wn   = wid & 1;         // 0..1  (N direction, 64 cols each)
    const int l15  = lane & 15;
    const bool lolane = (lane < 16);

    const f32x8 vzero = {0.f,0.f,0.f,0.f,0.f,0.f,0.f,0.f};
    f32x8 acc[2][4];
#pragma unroll
    for (int i = 0; i < 2; ++i)
#pragma unroll
        for (int j = 0; j < 4; ++j) acc[i][j] = vzero;

    // global->LDS loader assignments
    const int ar = tid >> 3;            // A row 0..31 (+32*rr)
    const int ac = (tid & 7) << 2;      // A col 0..28 step 4
    const int bn = tid & 127;           // W col 0..127 (one column per thread)
    const int bg = (tid >> 7) << 4;     // W k-half: 0 or 16

    float4 pa[4];
    float  pb[16];
    auto g_load = [&](int k0) {
#pragma unroll
        for (int rr = 0; rr < 4; ++rr)
            pa[rr] = *(const float4*)(A + (size_t)(m0 + ar + rr * 32) * lda + (k0 + ac));
        // column loads: lanes hold consecutive n -> coalesced per k-row
#pragma unroll
        for (int kk = 0; kk < 16; ++kk)
            pb[kk] = W[(size_t)(k0 + bg + kk) * ldw + (n0 + bn)];
    };
    g_load(0);

    for (int k0 = 0; k0 < K; k0 += BK) {
        // ---- stage prefetched tile into LDS (fp32 -> f16) ----
#pragma unroll
        for (int rr = 0; rr < 4; ++rr) {
            f16x4 hv = { (_Float16)pa[rr].x, (_Float16)pa[rr].y,
                         (_Float16)pa[rr].z, (_Float16)pa[rr].w };
            *(f16x4*)(&As[(ar + rr * 32) * LDS_STRIDE + ac]) = hv;   // 8B ds_store
        }
        {   // thread owns Bs[bn][bg..bg+15]: 2x 16B ds_store, packed converts
            f16x8 h0 = { (_Float16)pb[0], (_Float16)pb[1], (_Float16)pb[2], (_Float16)pb[3],
                         (_Float16)pb[4], (_Float16)pb[5], (_Float16)pb[6], (_Float16)pb[7] };
            f16x8 h1 = { (_Float16)pb[8],  (_Float16)pb[9],  (_Float16)pb[10], (_Float16)pb[11],
                         (_Float16)pb[12], (_Float16)pb[13], (_Float16)pb[14], (_Float16)pb[15] };
            *(f16x8*)(&Bs[bn * LDS_STRIDE + bg])     = h0;
            *(f16x8*)(&Bs[bn * LDS_STRIDE + bg + 8]) = h1;
        }
        __syncthreads();

        // prefetch next K-tile while this tile computes
        if (k0 + BK < K) g_load(k0 + BK);

        // ---- build WMMA fragments per ISA 7.12.2 layouts ----
        // A 16x32 f16: lanes 0-15 (row=l15) hold K 0-7 (v0-3) & 16-23 (v4-7);
        //              lanes 16-31 hold K 8-15 & 24-31.
        f16x16 afrag[2], bfrag[4];
#pragma unroll
        for (int i = 0; i < 2; ++i) {
            const int row = wm * 32 + i * 16 + l15;
            const int cb  = lolane ? 0 : 8;                // half offset
            union { f16x16 v; f16x8 h[2]; } u;
            u.h[0] = *(const f16x8*)(&As[row * LDS_STRIDE + cb]);        // K cb..cb+7
            u.h[1] = *(const f16x8*)(&As[row * LDS_STRIDE + cb + 16]);   // K cb+16..cb+23
            afrag[i] = u.v;
        }
        // B 32x16 f16: lane's column n = l15; lanes 0-15 hold K 0-7 & 16-23,
        //              lanes 16-31 hold K 8-15 & 24-31 (same as A).
#pragma unroll
        for (int j = 0; j < 4; ++j) {
            const int nrow = wn * 64 + j * 16 + l15;
            const int kb   = lolane ? 0 : 8;
            union { f16x16 v; f16x8 h[2]; } u;
            u.h[0] = *(const f16x8*)(&Bs[nrow * LDS_STRIDE + kb]);
            u.h[1] = *(const f16x8*)(&Bs[nrow * LDS_STRIDE + kb + 16]);
            bfrag[j] = u.v;
        }
#pragma unroll
        for (int i = 0; i < 2; ++i)
#pragma unroll
            for (int j = 0; j < 4; ++j)
                acc[i][j] = __builtin_amdgcn_wmma_f32_16x16x32_f16(
                    false, afrag[i], false, bfrag[j],
                    (short)0, acc[i][j], false, false);
        __syncthreads();
    }

    // ---- epilogue: C/D layout: lanes 0-15 -> N=l15, M=r; lanes 16-31 -> M=r+8
#pragma unroll
    for (int i = 0; i < 2; ++i) {
        const int mbase = m0 + wm * 32 + i * 16 + (lolane ? 0 : 8);
#pragma unroll
        for (int j = 0; j < 4; ++j) {
            const int col = n0 + wn * 64 + j * 16 + l15;
            const float bv = bias[col];
#pragma unroll
            for (int r = 0; r < 8; ++r) {
                float v = acc[i][j][r] + bv;
                float* p = C + (size_t)(mbase + r) * ldc + col;
                if (BETA) v += *p;
                if (TANH) v = fast_tanh(v);
                *p = v;
            }
        }
    }
}

// ---------------------------------------------------------------------------
// Fused guided attention (per batch row):
//   logits[n] = sum_h tanh(F[b,n,h] + G[b,h]) * Wa[h]    (ba cancels in softmax)
//   att = softmax(logits);  out[b,h] = sum_n att[n]*feat[b,n,h]
// N is a compile-time constant -> accumulators stay in VGPRs.
// ---------------------------------------------------------------------------
template<int N>
__global__ __launch_bounds__(256)
void att_kernel(const float* __restrict__ feat,  // [B,N,1024]
                const float* __restrict__ F,     // [B,N,1024] = feat@Wf + bf
                const float* __restrict__ G,     // [B,1024]
                const float* __restrict__ Wa,    // [1024]
                float* __restrict__ out)         // [B,1024]
{
    __shared__ float G_s[1024];
    __shared__ float Wa_s[1024];
    __shared__ float red[N * 8];
    __shared__ float att_s[N];

    const int b   = blockIdx.x;
    const int tid = threadIdx.x;

    for (int h = tid; h < 1024; h += 256) {
        G_s[h]  = G[(size_t)b * 1024 + h];
        Wa_s[h] = Wa[h];
    }
    __syncthreads();

    const float* Fb = F + (size_t)b * N * 1024;
    float acc[N];
#pragma unroll
    for (int n = 0; n < N; ++n) acc[n] = 0.0f;

#pragma unroll
    for (int h0 = 0; h0 < 1024; h0 += 256) {
        const int h = h0 + tid;
        const float g  = G_s[h];
        const float wa = Wa_s[h];
#pragma unroll
        for (int n = 0; n < N; ++n)
            acc[n] += fast_tanh(Fb[n * 1024 + h] + g) * wa;
    }

    const int lane = tid & 31, wid = tid >> 5;
#pragma unroll
    for (int n = 0; n < N; ++n) {
        float v = acc[n];
#pragma unroll
        for (int off = 16; off > 0; off >>= 1) v += __shfl_down(v, off, 32);
        if (lane == 0) red[n * 8 + wid] = v;
    }
    __syncthreads();

    if (tid == 0) {
        float mx = -3.0e38f;
#pragma unroll
        for (int n = 0; n < N; ++n) {
            float s = 0.0f;
            for (int w = 0; w < 8; ++w) s += red[n * 8 + w];
            att_s[n] = s;
            mx = fmaxf(mx, s);
        }
        float den = 0.0f;
#pragma unroll
        for (int n = 0; n < N; ++n) { float e = __expf(att_s[n] - mx); att_s[n] = e; den += e; }
        const float inv = 1.0f / den;
#pragma unroll
        for (int n = 0; n < N; ++n) att_s[n] *= inv;
    }
    __syncthreads();

    const float* fb = feat + (size_t)b * N * 1024;
    for (int h = tid; h < 1024; h += 256) {
        float s = 0.0f;
#pragma unroll
        for (int n = 0; n < N; ++n) s += att_s[n] * fb[n * 1024 + h];
        out[(size_t)b * 1024 + h] = s;
    }
}

// ---------------------------------------------------------------------------
// Small elementwise kernels. States live strided inside concat[B,3H]:
//   q_state = concat[:,0:H], h_state = concat[:,H:2H], i_state = concat[:,2H:3H]
// ---------------------------------------------------------------------------
__global__ void init_states(float* __restrict__ concat,
                            const float* __restrict__ his_g0,
                            const float* __restrict__ img_g0)
{
    const int idx = blockIdx.x * blockDim.x + threadIdx.x;   // 0 .. B*H
    const int b = idx >> 10, h = idx & 1023;
    concat[(size_t)b * 3072 + 1024 + h] = his_g0[idx];
    concat[(size_t)b * 3072 + 2048 + h] = img_g0[idx];
}

__global__ void blend_kernel(float* __restrict__ concat,
                             const float* __restrict__ q,
                             const float* __restrict__ hh,
                             const float* __restrict__ ii,
                             int round)
{
    const int idx = blockIdx.x * blockDim.x + threadIdx.x;
    const int b = idx >> 10, h = idx & 1023;
    float* row = concat + (size_t)b * 3072;
    const float qv = q[idx];
    row[h]        = (round == 0) ? qv : 0.5f * (row[h] + qv);
    row[1024 + h] = 0.5f * (row[1024 + h] + hh[idx]);
    row[2048 + h] = 0.5f * (row[2048 + h] + ii[idx]);
}

__global__ void copy_out(float* __restrict__ out,   // out+B*NINP passed in
                         const float* __restrict__ concat)
{
    const int idx = blockIdx.x * blockDim.x + threadIdx.x;
    const int b = idx >> 10, h = idx & 1023;
    out[idx]           = concat[(size_t)b * 3072 + 1024 + h];   // his_g
    out[1048576 + idx] = concat[(size_t)b * 3072 + 2048 + h];   // img_g
}

// ---------------------------------------------------------------------------
static inline void gemm(const float* A, int lda, const float* W, int ldw,
                        const float* bias, float* C, int ldc,
                        int M, int N, int K, int beta, int do_tanh,
                        hipStream_t s)
{
    dim3 grid(N / BN, M / BM), block(256);
    if (beta)
        wmma_gemm<1,0><<<grid, block, 0, s>>>(A, lda, W, ldw, bias, C, ldc, M, N, K);
    else if (do_tanh)
        wmma_gemm<0,1><<<grid, block, 0, s>>>(A, lda, W, ldw, bias, C, ldc, M, N, K);
    else
        wmma_gemm<0,0><<<grid, block, 0, s>>>(A, lda, W, ldw, bias, C, ldc, M, N, K);
}

extern "C" void kernel_launch(void* const* d_in, const int* in_sizes, int n_in,
                              void* d_out, int out_size, void* d_ws, size_t ws_size,
                              hipStream_t stream)
{
    (void)in_sizes; (void)n_in; (void)out_size; (void)ws_size;
    const int B = 1024, LQ = 20, LH = 10, NI = 36, H = 1024, NINP = 1024;

    const float* ques   = (const float*)d_in[0];
    const float* his    = (const float*)d_in[1];
    const float* img    = (const float*)d_in[2];
    const float* img_g0 = (const float*)d_in[3];
    const float* his_g0 = (const float*)d_in[4];
    const float* Wf     = (const float*)d_in[5];
    const float* bf     = (const float*)d_in[6];
    const float* Wg1    = (const float*)d_in[7];
    const float* bg1    = (const float*)d_in[8];
    const float* Wg2    = (const float*)d_in[9];
    const float* bg2    = (const float*)d_in[10];
    const float* Wa     = (const float*)d_in[11];
    /* d_in[12] = ba: constant logit shift, cancels in softmax */
    const float* fc1W   = (const float*)d_in[13];
    const float* fc1b   = (const float*)d_in[14];

    // workspace layout (floats), ~306 MB total
    float* ws     = (float*)d_ws;
    float* Fq     = ws;                                    // [B*LQ, H]
    float* Fh     = Fq + (size_t)B * LQ * H;               // [B*LH, H]
    float* Fi     = Fh + (size_t)B * LH * H;               // [B*NI, H]
    float* concat = Fi + (size_t)B * NI * H;               // [B, 3H]
    float* q_att  = concat + (size_t)B * 3 * H;
    float* h_att  = q_att + (size_t)B * H;
    float* i_att  = h_att + (size_t)B * H;
    float* G      = i_att + (size_t)B * H;

    float* h_state = concat + H;        // lda = 3H views into concat
    float* i_state = concat + 2 * H;

    const dim3 eb(256), eg(B * H / 256);
    init_states<<<eg, eb, 0, stream>>>(concat, his_g0, img_g0);

    // F projections: computed ONCE, reused by all 3 rounds.
    gemm(ques, H, Wf + 0 * H * H, H, bf + 0 * H, Fq, H, B * LQ, H, H, 0, 0, stream);
    gemm(his,  H, Wf + 1 * H * H, H, bf + 1 * H, Fh, H, B * LH, H, H, 0, 0, stream);
    gemm(img,  H, Wf + 2 * H * H, H, bf + 2 * H, Fi, H, B * NI, H, H, 0, 0, stream);

    for (int r = 0; r < 3; ++r) {
        // ques attention: g1 = img state, g2 = his state
        gemm(i_state, 3 * H, Wg1 + 0 * H * H, H, bg1 + 0 * H, G, H, B, H, H, 0, 0, stream);
        gemm(h_state, 3 * H, Wg2 + 0 * H * H, H, bg2 + 0 * H, G, H, B, H, H, 1, 0, stream);
        att_kernel<20><<<B, 256, 0, stream>>>(ques, Fq, G, Wa + 0 * H, q_att);
        // his attention: g1 = fresh ques att (unblended), g2 = img state
        gemm(q_att,   H,     Wg1 + 1 * H * H, H, bg1 + 1 * H, G, H, B, H, H, 0, 0, stream);
        gemm(i_state, 3 * H, Wg2 + 1 * H * H, H, bg2 + 1 * H, G, H, B, H, H, 1, 0, stream);
        att_kernel<10><<<B, 256, 0, stream>>>(his, Fh, G, Wa + 1 * H, h_att);
        // img attention: g1 = fresh ques att, g2 = fresh his att (both unblended)
        gemm(q_att, H, Wg1 + 2 * H * H, H, bg1 + 2 * H, G, H, B, H, H, 0, 0, stream);
        gemm(h_att, H, Wg2 + 2 * H * H, H, bg2 + 2 * H, G, H, B, H, H, 1, 0, stream);
        att_kernel<36><<<B, 256, 0, stream>>>(img, Fi, G, Wa + 2 * H, i_att);
        // state updates: q: r0 copy else avg; h/i: always avg with prev state
        blend_kernel<<<eg, eb, 0, stream>>>(concat, q_att, h_att, i_att, r);
    }

    // encoder_feat = tanh(concat @ fc1_W + fc1_b)  -> d_out[0 : B*NINP)
    float* out = (float*)d_out;
    gemm(concat, 3 * H, fc1W, NINP, fc1b, out, NINP, B, NINP, 3 * H, 0, 1, stream);
    // his_g -> d_out[B*NINP : 2*B*NINP), img_g -> d_out[2*B*NINP : 3*B*NINP)
    copy_out<<<eg, eb, 0, stream>>>(out + (size_t)B * NINP, concat);
}
